// TimeAttention_4449586118922
// MI455X (gfx1250) — matrix operand order repaired and checked by reading, not compiled, not run on hardware
//
#include <hip/hip_runtime.h>
#include <hip/hip_bf16.h>

typedef __attribute__((ext_vector_type(16))) _Float16 v16h;
typedef __attribute__((ext_vector_type(8)))  _Float16 v8h;
typedef __attribute__((ext_vector_type(8)))  float    v8f;

#define NB_   8
#define NV_   7
#define NT_   128
#define NH_   8
#define NE_   64
#define NL_   (NV_ * NT_)   // 896
#define DIMR_ 512

// ln(10000)/16  (theta_j = 10000^(-2j/32) = exp(-j * ln(10000)/16))
#define ROPE_C 0.5756462732485114f

__device__ __forceinline__ float rope_theta(int j) {
    return __expf(-ROPE_C * (float)j);
}

// ---------------------------------------------------------------------------
// Kernel 1: 7x7 var-similarity matrix S = triu(cos(R1),0) + tril(cos(R2),-1)
// ---------------------------------------------------------------------------
__global__ void sim_kernel(const float* __restrict__ R1,
                           const float* __restrict__ R2,
                           float* __restrict__ S) {
    int t = threadIdx.x;
    if (t >= NV_ * NV_) return;
    int i = t / NV_, j = t % NV_;
    const float* R = (i <= j) ? R1 : R2;
    const float* a = R + i * DIMR_;
    const float* b = R + j * DIMR_;
    float dab = 0.f, daa = 0.f, dbb = 0.f;
    for (int q = 0; q < DIMR_; ++q) {
        float x = a[q], y = b[q];
        dab += x * y; daa += x * x; dbb += y * y;
    }
    float na = fmaxf(sqrtf(daa), 1e-12f);
    float nb = fmaxf(sqrtf(dbb), 1e-12f);
    S[t] = dab / (na * nb);
}

// ---------------------------------------------------------------------------
// Kernel 2: fused RoPE + attention (flash style), one 16-row q-tile per wave.
// K and V chunks are cooperatively staged in LDS (K with RoPE pre-applied),
// so the inner loop is pure ds_load_b128 -> v_wmma.
// ---------------------------------------------------------------------------
__global__ void __launch_bounds__(256)
attn_kernel(const float* __restrict__ Q, const float* __restrict__ K,
            const float* __restrict__ V, const float* __restrict__ bias_emb,
            const int* __restrict__ date, const float* __restrict__ Smat,
            float* __restrict__ Out) {
    __shared__ alignas(32) _Float16 kLDS[32][NE_];      // rope'd K chunk [k][e], 4KB
    __shared__ alignas(32) _Float16 vLDS[NE_][32];      // V chunk       [e][k], 4KB
    __shared__ alignas(32) _Float16 pLDS[8][16][32];    // per-wave P staging, 8KB

    const int b    = blockIdx.z;
    const int h    = blockIdx.y;
    const int tile = blockIdx.x;          // query var index (16 | 128)
    const int tid  = threadIdx.x;
    const int w    = tid >> 5;
    const int lane = tid & 31;
    const int lane16 = lane & 15;
    const int hi   = lane >> 4;

    const int    qm0   = tile * 128 + w * 16;
    const int    var_m = tile;
    const float  scale = 0.125f;          // 1/sqrt(64)
    const size_t rowst = (size_t)NH_ * NE_;

    const float* Kb = K + (size_t)b * NL_ * rowst + (size_t)h * NE_;
    const float* Vb = V + (size_t)b * NL_ * rowst + (size_t)h * NE_;

    // ---- Q tile -> WMMA A layout (f16) with fused RoPE ----
    const int    mrow = qm0 + lane16;
    const float* qp   = Q + ((size_t)b * NL_ + mrow) * rowst + (size_t)h * NE_;
    const float  sidq = (float)date[b * NT_ + (mrow % NT_)];
    const int    kb8  = hi * 8;
    v16h aLo, aHi;
#pragma unroll
    for (int g = 0; g < 2; ++g) {
        const int ks = g * 16 + kb8;
#pragma unroll
        for (int p = 0; p < 4; ++p) {             // K = ks..ks+7 (< 32) -> RoPE
            int kk = ks + 2 * p;
            float x0 = qp[kk], x1 = qp[kk + 1];
            float ang = sidq * rope_theta(kk >> 1);
            float c = __cosf(ang), s = __sinf(ang);
            aLo[g * 8 + 2 * p]     = (_Float16)(x0 * c - x1 * s);
            aLo[g * 8 + 2 * p + 1] = (_Float16)(x1 * c + x0 * s);
        }
#pragma unroll
        for (int i = 0; i < 8; ++i)               // K = 32+ks.. : no RoPE
            aHi[g * 8 + i] = (_Float16)qp[32 + ks + i];
    }

    // ---- online softmax state ----
    float rmax[8], rsum[8];
    v8f acc[4] = {{}, {}, {}, {}};
#pragma unroll
    for (int r = 0; r < 8; ++r) { rmax[r] = -1e30f; rsum[r] = 0.f; }

    for (int chunk = 0; chunk < 28; ++chunk) {
        const int s0c  = chunk * 32;
        const int tokb = (chunk & 3) * 32;        // key token base within var

        // ---- cooperative stage: K (rope'd, [k][e]) and V ([e][k]) -> LDS f16 ----
        {
            const int kk = tid >> 3;              // 0..31 : key row in chunk
            const int eb = (tid & 7) * 8;         // 0..56 : element base
            const float* kp = Kb + (size_t)(s0c + kk) * rowst + eb;
            const float* vp = Vb + (size_t)(s0c + kk) * rowst + eb;
#pragma unroll
            for (int i = 0; i < 8; ++i) vLDS[eb + i][kk] = (_Float16)vp[i];
            if (eb < 32) {                        // RoPE region
                const float sidk = (float)date[b * NT_ + tokb + kk];
#pragma unroll
                for (int p = 0; p < 4; ++p) {
                    float x0 = kp[2 * p], x1 = kp[2 * p + 1];
                    float ang = sidk * rope_theta((eb >> 1) + p);
                    float c = __cosf(ang), s = __sinf(ang);
                    kLDS[kk][eb + 2 * p]     = (_Float16)(x0 * c - x1 * s);
                    kLDS[kk][eb + 2 * p + 1] = (_Float16)(x1 * c + x0 * s);
                }
            } else {
#pragma unroll
                for (int i = 0; i < 8; ++i) kLDS[kk][eb + i] = (_Float16)kp[i];
            }
            if (chunk < 27) {                     // hide next chunk's latency
                __builtin_prefetch(kp + 32 * rowst, 0, 0);
                __builtin_prefetch(vp + 32 * rowst, 0, 0);
            }
        }
        __syncthreads();

        if (tokb <= (w * 16 + 15)) {              // wave-uniform: skip fully-masked
            const int   var_n = chunk >> 2;
            const float biasv = bias_emb[(var_m == var_n ? NH_ : 0) + h];
            const float sfac  = Smat[var_m * NV_ + var_n];
            const int   kb16  = hi * 16;

            // ---- two 16-key score sub-tiles: B layout straight from kLDS ----
            float sA[2][8];
#pragma unroll
            for (int st = 0; st < 2; ++st) {
                const int nl = st * 16 + lane16;  // local key row
                v8h b0 = *(const v8h*)&kLDS[nl][kb8];
                v8h b1 = *(const v8h*)&kLDS[nl][16 + kb8];
                v8h b2 = *(const v8h*)&kLDS[nl][32 + kb8];
                v8h b3 = *(const v8h*)&kLDS[nl][48 + kb8];
                v16h bLo, bHi;
#pragma unroll
                for (int i = 0; i < 8; ++i) {
                    bLo[i] = b0[i]; bLo[8 + i] = b1[i];
                    bHi[i] = b2[i]; bHi[8 + i] = b3[i];
                }
                v8f c = {};
                c = __builtin_amdgcn_wmma_f32_16x16x32_f16(false, aLo, false, bLo, (short)0, c, false, false);
                c = __builtin_amdgcn_wmma_f32_16x16x32_f16(false, aHi, false, bHi, (short)0, c, false, false);
                const int tn = tokb + st * 16 + lane16;
#pragma unroll
                for (int r = 0; r < 8; ++r) {
                    const int tokm = w * 16 + r + hi * 8;
                    float s = scale * (c[r] + biasv);
                    sA[st][r] = (tn > tokm) ? -1e30f : s;
                }
            }

            // ---- online softmax update (row reduce across 16-lane halves) ----
            float tmax[8], psum[8];
#pragma unroll
            for (int r = 0; r < 8; ++r) tmax[r] = fmaxf(sA[0][r], sA[1][r]);
#pragma unroll
            for (int d = 1; d < 16; d <<= 1)
#pragma unroll
                for (int r = 0; r < 8; ++r)
                    tmax[r] = fmaxf(tmax[r], __shfl_xor(tmax[r], d, 32));
#pragma unroll
            for (int r = 0; r < 8; ++r) {
                float nm = fmaxf(rmax[r], tmax[r]);
                float f  = __expf(rmax[r] - nm);
                rmax[r]  = nm;
                float p0 = __expf(sA[0][r] - nm);
                float p1 = __expf(sA[1][r] - nm);
                sA[0][r] = p0; sA[1][r] = p1;
                psum[r]  = p0 + p1;
                rsum[r] *= f;
#pragma unroll
                for (int t = 0; t < 4; ++t) acc[t][r] *= f;
            }
#pragma unroll
            for (int d = 1; d < 16; d <<= 1)
#pragma unroll
                for (int r = 0; r < 8; ++r)
                    psum[r] += __shfl_xor(psum[r], d, 32);
#pragma unroll
            for (int r = 0; r < 8; ++r) rsum[r] += psum[r];

            // ---- P (x sim) : C-layout -> A-layout via per-wave LDS ----
#pragma unroll
            for (int r = 0; r < 8; ++r) {
                const int M = r + hi * 8;
                pLDS[w][M][lane16]      = (_Float16)(sA[0][r] * sfac);
                pLDS[w][M][16 + lane16] = (_Float16)(sA[1][r] * sfac);
            }
            __asm__ volatile("s_wait_dscnt 0" ::: "memory");  // wave-internal LDS RAW
            const int pkb = hi * 8;
            v8h p0v = *(const v8h*)&pLDS[w][lane16][pkb];
            v8h p1v = *(const v8h*)&pLDS[w][lane16][16 + pkb];
            v16h pA;
#pragma unroll
            for (int i = 0; i < 8; ++i) { pA[i] = p0v[i]; pA[8 + i] = p1v[i]; }

            // ---- O += P(16x32) x V(32x16) for each of 4 e-chunks ----
#pragma unroll
            for (int t = 0; t < 4; ++t) {
                const int e = t * 16 + lane16;
                v8h l8 = *(const v8h*)&vLDS[e][pkb];
                v8h h8 = *(const v8h*)&vLDS[e][16 + pkb];
                v16h bV;
#pragma unroll
                for (int i = 0; i < 8; ++i) { bV[i] = l8[i]; bV[8 + i] = h8[i]; }
                acc[t] = __builtin_amdgcn_wmma_f32_16x16x32_f16(
                    false, pA, false, bV, (short)0, acc[t], false, false);
            }
        }
        __syncthreads();
    }

    // ---- normalize and store: out[b, l, h, e] ----
#pragma unroll
    for (int r = 0; r < 8; ++r) {
        const float inv = 1.0f / rsum[r];
        const int   M   = r + hi * 8;
        float* op = Out + ((size_t)b * NL_ + (qm0 + M)) * rowst + (size_t)h * NE_ + lane16;
#pragma unroll
        for (int t = 0; t < 4; ++t) op[t * 16] = acc[t][r] * inv;
    }
}

extern "C" void kernel_launch(void* const* d_in, const int* in_sizes, int n_in,
                              void* d_out, int out_size, void* d_ws, size_t ws_size,
                              hipStream_t stream) {
    (void)in_sizes; (void)n_in; (void)out_size; (void)ws_size;
    const float* q        = (const float*)d_in[0];
    const float* k        = (const float*)d_in[1];
    const float* v        = (const float*)d_in[2];
    const float* bias_emb = (const float*)d_in[3];
    const float* R1       = (const float*)d_in[4];
    const float* R2       = (const float*)d_in[5];
    const int*   date     = (const int*)d_in[6];
    float*       Smat     = (float*)d_ws;          // 49 floats
    float*       out      = (float*)d_out;

    hipLaunchKernelGGL(sim_kernel, dim3(1), dim3(64), 0, stream, R1, R2, Smat);
    hipLaunchKernelGGL(attn_kernel, dim3(NL_ / 128, NH_, NB_), dim3(256), 0, stream,
                       q, k, v, bias_emb, date, Smat, out);
}
